// PointTransformerLayerScalar_75093208203621
// MI455X (gfx1250) — matrix operand order repaired and checked by reading, not compiled, not run on hardware
//
#include <hip/hip_runtime.h>
#include <math.h>

#define N_PTS 65536
#define C     64
#define NS    16
#define BN_EPS 1e-5f

typedef __attribute__((ext_vector_type(16))) _Float16 v16h;
typedef __attribute__((ext_vector_type(8)))  _Float16 v8h;
typedef __attribute__((ext_vector_type(4)))  _Float16 v4h;
typedef __attribute__((ext_vector_type(8)))  float    v8f;

// ---------------------------------------------------------------------------
// Kernel 0: zero the 8-float stats accumulator in workspace (harness poisons ws)
// ---------------------------------------------------------------------------
__global__ void pt_zero_stats(float* stats) {
    if (threadIdx.x < 8) stats[threadIdx.x] = 0.0f;
}

// ---------------------------------------------------------------------------
// Kernel 1: pre-convert X (N x 64, f32) to f16 once; shared by all 3 GEMMs.
// Each thread: one float4 load -> one 8-byte store.
// ---------------------------------------------------------------------------
__global__ __launch_bounds__(256) void pt_x_to_f16(
    const float* __restrict__ X, _Float16* __restrict__ Xh)
{
    const int t = blockIdx.x * blockDim.x + threadIdx.x;  // N*C/4 threads exact
    const float4 f = ((const float4*)X)[t];
    v4h h;
    h[0] = (_Float16)f.x; h[1] = (_Float16)f.y;
    h[2] = (_Float16)f.z; h[3] = (_Float16)f.w;
    ((v4h*)Xh)[t] = h;
}

// ---------------------------------------------------------------------------
// Kernel 2: BatchNorm statistics over h = (p[idx]-p) @ Wp1 + bp1  (N*NS x 3)
// wave shuffle pre-reduction -> LDS float atomics -> global float atomics
// ---------------------------------------------------------------------------
__global__ __launch_bounds__(256) void pt_stats(
    const float* __restrict__ p, const int* __restrict__ idx,
    const float* __restrict__ Wp1, const float* __restrict__ bp1,
    float* __restrict__ stats)
{
    __shared__ float s[6];
    if (threadIdx.x < 6) s[threadIdx.x] = 0.0f;
    __syncthreads();

    const int gid = blockIdx.x * blockDim.x + threadIdx.x;   // exact: N*NS threads
    const int i = gid >> 4;
    const int j = gid & 15;
    const int nb = idx[i * NS + j];

    const float pr0 = p[nb * 3 + 0] - p[i * 3 + 0];
    const float pr1 = p[nb * 3 + 1] - p[i * 3 + 1];
    const float pr2 = p[nb * 3 + 2] - p[i * 3 + 2];

    #pragma unroll
    for (int d = 0; d < 3; ++d) {
        float h  = pr0 * Wp1[0 * 3 + d] + pr1 * Wp1[1 * 3 + d] + pr2 * Wp1[2 * 3 + d] + bp1[d];
        float h2 = h * h;
        #pragma unroll
        for (int o = 16; o > 0; o >>= 1) {
            h  += __shfl_xor(h,  o, 32);
            h2 += __shfl_xor(h2, o, 32);
        }
        if ((threadIdx.x & 31) == 0) {
            atomicAdd(&s[d],     h);
            atomicAdd(&s[3 + d], h2);
        }
    }
    __syncthreads();
    if (threadIdx.x < 6) atomicAdd(&stats[threadIdx.x], s[threadIdx.x]);
}

// ---------------------------------------------------------------------------
// Kernel 3: Y = X @ W + b  (N x 64 @ 64 x 64) via v_wmma_f32_16x16x32_f16
// W staged in LDS as f16 TRANSPOSED (n-major) -> B fragment = 2x ds_load_b128.
// A read from pre-converted f16 X      -> A fragment = 2x global_load_b128.
// One wave computes a 16x64 tile with 4 f32 accumulators, 2 K-steps.
// ---------------------------------------------------------------------------
__global__ __launch_bounds__(256) void pt_qkv_gemm(
    const _Float16* __restrict__ Xh, const float* __restrict__ W,
    const float* __restrict__ bias, float* __restrict__ Y)
{
    __shared__ _Float16 sWt[C * C];   // 8 KB, sWt[n*C + k] = (f16)W[k*C + n]
    __shared__ float    sB[C];

    // stage W transposed + converted (one time; global reads hit cache, LDS
    // writes are contiguous 2B per lane)
    for (int t = threadIdx.x; t < C * C; t += 256) {
        const int n = t >> 6, k = t & 63;
        sWt[n * C + k] = (_Float16)W[k * C + n];
    }
    if (threadIdx.x < C) sB[threadIdx.x] = bias[threadIdx.x];
    __syncthreads();

    const int wave = threadIdx.x >> 5;
    const int lane = threadIdx.x & 31;
    const int hi   = lane >> 4;          // half-wave select
    const int l15  = lane & 15;
    const int row0 = (blockIdx.x * 8 + wave) * 16;
    const int row  = row0 + l15;

    v8f acc[4] = {};                      // 4 n-tiles of 16x16 f32

    #pragma unroll
    for (int kb = 0; kb < C; kb += 32) {
        // ----- A fragment (16x32 f16): lane holds row `row`;
        //       halfs 0-7 -> K=koff..koff+7, halfs 8-15 -> K=koff+16..koff+23
        const int koff = kb + hi * 8;
        const v8h* xa = (const v8h*)(Xh + (size_t)row * C + koff);
        const v8h alo = xa[0];            // K = koff .. koff+7
        const v8h ahi = xa[2];            // K = koff+16 .. koff+23
        const v16h a = __builtin_shufflevector(alo, ahi,
            0,1,2,3,4,5,6,7,8,9,10,11,12,13,14,15);

        const int kB = kb + hi * 8;       // B half-wave K offset
        #pragma unroll
        for (int nt = 0; nt < 4; ++nt) {
            // ----- B fragment (32x16 f16): lane holds column nt*16+l15,
            //       halfs 0-7 -> K=kB..kB+7, halfs 8-15 -> K=kB+16..kB+23
            const v8h* wp = (const v8h*)(sWt + (nt * 16 + l15) * C + kB);
            const v16h b = __builtin_shufflevector(wp[0], wp[2],
                0,1,2,3,4,5,6,7,8,9,10,11,12,13,14,15);

            acc[nt] = __builtin_amdgcn_wmma_f32_16x16x32_f16(
                false, a, false, b, (short)0, acc[nt], false, false);
        }
    }

    // ----- epilogue: D layout is VGPR r -> rows {r, r+8}, lanes -> columns
    #pragma unroll
    for (int nt = 0; nt < 4; ++nt) {
        const int col = nt * 16 + l15;
        const float bv = sB[col];
        #pragma unroll
        for (int r = 0; r < 8; ++r) {
            Y[(size_t)(row0 + r + 8 * hi) * C + col] = acc[nt][r] + bv;
        }
    }
}

// ---------------------------------------------------------------------------
// Kernel 4: fused gather + positional embedding + attention softmax + output
// One wave per point; lane handles channels (lane, lane+32) -> coalesced rows.
// ---------------------------------------------------------------------------
__global__ __launch_bounds__(256) void pt_attn(
    const float* __restrict__ p, const int* __restrict__ idx,
    const float* __restrict__ Qm, const float* __restrict__ Km,
    const float* __restrict__ Vm, const float* __restrict__ stats,
    const float* __restrict__ Wp1, const float* __restrict__ bp1,
    const float* __restrict__ gamma, const float* __restrict__ beta,
    const float* __restrict__ Wp2, const float* __restrict__ bp2,
    float* __restrict__ out)
{
    const int wave = threadIdx.x >> 5;
    const int lane = threadIdx.x & 31;
    const int i    = blockIdx.x * 8 + wave;
    const int c1   = lane;
    const int c2   = lane + 32;

    // BatchNorm affine folded: h*scale + shift
    const float inv_cnt = 1.0f / (float)(N_PTS * NS);
    float scale[3], shift[3];
    #pragma unroll
    for (int d = 0; d < 3; ++d) {
        float mean = stats[d] * inv_cnt;
        float var  = stats[3 + d] * inv_cnt - mean * mean;
        float sc   = rsqrtf(var + BN_EPS) * gamma[d];
        scale[d] = sc;
        shift[d] = beta[d] - mean * sc;
    }

    float w1[9], b1[3];
    #pragma unroll
    for (int t = 0; t < 9; ++t) w1[t] = Wp1[t];
    #pragma unroll
    for (int d = 0; d < 3; ++d) b1[d] = bp1[d];
    float w2a[3], w2b[3];
    #pragma unroll
    for (int d = 0; d < 3; ++d) { w2a[d] = Wp2[d * C + c1]; w2b[d] = Wp2[d * C + c2]; }
    const float bpa = bp2[c1], bpb = bp2[c2];

    const float q1 = Qm[(size_t)i * C + c1];
    const float q2 = Qm[(size_t)i * C + c2];
    const float pi0 = p[i * 3 + 0], pi1 = p[i * 3 + 1], pi2 = p[i * 3 + 2];

    float logit[NS], vv1[NS], vv2[NS];

    #pragma unroll
    for (int j = 0; j < NS; ++j) {
        const int nb = idx[i * NS + j];
        const float pr0 = p[nb * 3 + 0] - pi0;
        const float pr1 = p[nb * 3 + 1] - pi1;
        const float pr2 = p[nb * 3 + 2] - pi2;

        float hb[3];
        #pragma unroll
        for (int d = 0; d < 3; ++d) {
            float h = pr0 * w1[0 * 3 + d] + pr1 * w1[1 * 3 + d] + pr2 * w1[2 * 3 + d] + b1[d];
            h = h * scale[d] + shift[d];
            hb[d] = fmaxf(h, 0.0f);
        }
        const float pe1 = hb[0] * w2a[0] + hb[1] * w2a[1] + hb[2] * w2a[2] + bpa;
        const float pe2 = hb[0] * w2b[0] + hb[1] * w2b[1] + hb[2] * w2b[2] + bpb;

        const float k1 = Km[(size_t)nb * C + c1] + pe1;
        const float k2 = Km[(size_t)nb * C + c2] + pe2;
        vv1[j] = Vm[(size_t)nb * C + c1] + pe1;
        vv2[j] = Vm[(size_t)nb * C + c2] + pe2;

        float part = q1 * k1 + q2 * k2;
        #pragma unroll
        for (int o = 16; o > 0; o >>= 1) part += __shfl_xor(part, o, 32);
        logit[j] = part * 0.125f;              // / sqrt(64)
    }

    float m = logit[0];
    #pragma unroll
    for (int j = 1; j < NS; ++j) m = fmaxf(m, logit[j]);

    float ssum = 0.0f, o1 = 0.0f, o2 = 0.0f;
    #pragma unroll
    for (int j = 0; j < NS; ++j) {
        const float e = __expf(logit[j] - m);
        ssum += e;
        o1 += e * vv1[j];
        o2 += e * vv2[j];
    }
    const float inv = 1.0f / ssum;
    out[(size_t)i * C + c1] = o1 * inv;
    out[(size_t)i * C + c2] = o2 * inv;
}

// ---------------------------------------------------------------------------
extern "C" void kernel_launch(void* const* d_in, const int* in_sizes, int n_in,
                              void* d_out, int out_size, void* d_ws, size_t ws_size,
                              hipStream_t stream) {
    const float* p    = (const float*)d_in[0];
    const float* x    = (const float*)d_in[1];
    const int*   idx  = (const int*)  d_in[2];
    const float* Wq   = (const float*)d_in[3];
    const float* bq   = (const float*)d_in[4];
    const float* Wk   = (const float*)d_in[5];
    const float* bk   = (const float*)d_in[6];
    const float* Wv   = (const float*)d_in[7];
    const float* bv   = (const float*)d_in[8];
    const float* Wp1  = (const float*)d_in[9];
    const float* bp1  = (const float*)d_in[10];
    const float* gam  = (const float*)d_in[11];
    const float* bet  = (const float*)d_in[12];
    const float* Wp2  = (const float*)d_in[13];
    const float* bp2  = (const float*)d_in[14];
    float* out = (float*)d_out;

    float* ws    = (float*)d_ws;
    float* stats = ws;                           // 8 floats (sum[3], sumsq[3], pad)
    float* Qm    = ws + 16;                      // N*64 f32
    float* Km    = Qm + (size_t)N_PTS * C;       // N*64 f32
    float* Vm    = Km + (size_t)N_PTS * C;       // N*64 f32
    _Float16* Xh = (_Float16*)(Vm + (size_t)N_PTS * C);  // N*64 f16 (total ~58.7 MB)

    pt_zero_stats<<<1, 32, 0, stream>>>(stats);
    pt_x_to_f16<<<(N_PTS * C / 4) / 256, 256, 0, stream>>>(x, Xh);
    pt_stats<<<(N_PTS * NS) / 256, 256, 0, stream>>>(p, idx, Wp1, bp1, stats);

    const int gemm_blocks = N_PTS / (16 * 8);    // 512 blocks, 8 waves, 16 rows/wave
    pt_qkv_gemm<<<gemm_blocks, 256, 0, stream>>>(Xh, Wq, bq, Qm);
    pt_qkv_gemm<<<gemm_blocks, 256, 0, stream>>>(Xh, Wk, bk, Km);
    pt_qkv_gemm<<<gemm_blocks, 256, 0, stream>>>(Xh, Wv, bv, Vm);

    pt_attn<<<N_PTS / 8, 256, 0, stream>>>(p, idx, Qm, Km, Vm, stats,
                                           Wp1, bp1, gam, bet, Wp2, bp2, out);
}
